// PLinear_25451976196807
// MI455X (gfx1250) — matrix operand order repaired and checked by reading, not compiled, not run on hardware
//
#include <hip/hip_runtime.h>

// Problem constants (from reference)
#define TOKENS 8192
#define D_IN   4096
#define D_OUT  4096

// GEMM tiling: block tile 256x128, 8 waves (4x2), wave tile 64x64, K-step 32.
#define BM 256
#define BN 128
#define KB 32
#define LDA (KB + 8)   // padded LDS row stride (elements) -> 80B, 16B aligned
#define LDB (KB + 8)

typedef __attribute__((ext_vector_type(16))) __bf16 v16bf;
typedef __attribute__((ext_vector_type(8)))  __bf16 v8bf;
typedef __attribute__((ext_vector_type(8)))  float  v8f;
typedef __attribute__((ext_vector_type(4)))  int    v4i;

// gfx1250 async global->LDS copy (ASYNCcnt-tracked), guarded so the file
// still compiles on toolchains that don't declare the builtins.
#if defined(__has_builtin)
#  if __has_builtin(__builtin_amdgcn_global_load_async_to_lds_b128) && \
      __has_builtin(__builtin_amdgcn_s_wait_asynccnt)
#    define USE_ASYNC 1
#  endif
#endif
#ifndef USE_ASYNC
#  define USE_ASYNC 0
#endif

#if USE_ASYNC
// Builtin signature (from clang diagnostic): (v4i AS1*, v4i AS3*, imm, imm)
typedef __attribute__((address_space(1))) v4i gv4i;  // global int4
typedef __attribute__((address_space(3))) v4i lv4i;  // LDS int4
#endif

// ---------------- Pass 1a: x (f32) -> xb (bf16) ----------------
__global__ __launch_bounds__(256) void cvt_x_kernel(const float* __restrict__ x,
                                                    __bf16* __restrict__ xb) {
  size_t c = (size_t)blockIdx.x * blockDim.x + threadIdx.x;  // 8-element chunk
  const float4* x4 = (const float4*)x;
  float4 a = x4[2 * c];
  float4 b = x4[2 * c + 1];
  v8bf o;
  o[0] = (__bf16)a.x; o[1] = (__bf16)a.y; o[2] = (__bf16)a.z; o[3] = (__bf16)a.w;
  o[4] = (__bf16)b.x; o[5] = (__bf16)b.y; o[6] = (__bf16)b.z; o[7] = (__bf16)b.w;
  *(v8bf*)(xb + 8 * c) = o;
}

// ------- Pass 1b: W = (w_pos>0) - (w_neg>0) in {-1,0,1} as bf16 -------
__device__ __forceinline__ __bf16 tern(float p, float n) {
  return (__bf16)((p > 0.0f ? 1.0f : 0.0f) - (n > 0.0f ? 1.0f : 0.0f));
}

__global__ __launch_bounds__(256) void ternarize_kernel(const float* __restrict__ wp,
                                                        const float* __restrict__ wn,
                                                        __bf16* __restrict__ wb) {
  size_t c = (size_t)blockIdx.x * blockDim.x + threadIdx.x;  // 8-element chunk
  const float4* p4 = (const float4*)wp;
  const float4* n4 = (const float4*)wn;
  float4 p0 = p4[2 * c], p1 = p4[2 * c + 1];
  float4 q0 = n4[2 * c], q1 = n4[2 * c + 1];
  v8bf o;
  o[0] = tern(p0.x, q0.x); o[1] = tern(p0.y, q0.y);
  o[2] = tern(p0.z, q0.z); o[3] = tern(p0.w, q0.w);
  o[4] = tern(p1.x, q1.x); o[5] = tern(p1.y, q1.y);
  o[6] = tern(p1.z, q1.z); o[7] = tern(p1.w, q1.w);
  *(v8bf*)(wb + 8 * c) = o;
}

// ------------- staging: one K-slice of A (256x32) and B (128x32) -------------
__device__ __forceinline__ void stage_tiles(const __bf16* __restrict__ xb,
                                            const __bf16* __restrict__ wb,
                                            __bf16* As, __bf16* Bs,
                                            int tm0, int tn0, int kt, int tid) {
#pragma unroll
  for (int s = 0; s < 4; ++s) {  // A: 1024 16B chunks, 4 per thread
    int c  = tid + s * 256;
    int m  = c >> 2;
    int kc = (c & 3) << 3;
    const __bf16* g = &xb[(size_t)(tm0 + m) * D_IN + kt + kc];
    __bf16* l = &As[m * LDA + kc];
#if USE_ASYNC
    __builtin_amdgcn_global_load_async_to_lds_b128((gv4i*)g, (lv4i*)l, 0, 0);
#else
    *(v8bf*)l = *(const v8bf*)g;
#endif
  }
#pragma unroll
  for (int s = 0; s < 2; ++s) {  // B: 512 16B chunks, 2 per thread ([N][K])
    int c  = tid + s * 256;
    int n  = c >> 2;
    int kc = (c & 3) << 3;
    const __bf16* g = &wb[(size_t)(tn0 + n) * D_IN + kt + kc];
    __bf16* l = &Bs[n * LDB + kc];
#if USE_ASYNC
    __builtin_amdgcn_global_load_async_to_lds_b128((gv4i*)g, (lv4i*)l, 0, 0);
#else
    *(v8bf*)l = *(const v8bf*)g;
#endif
  }
}

// ---------------- Pass 2: y = xb @ Wᵀ via bf16 WMMA ----------------
__global__ __launch_bounds__(256) void tern_gemm_kernel(const __bf16* __restrict__ xb,
                                                        const __bf16* __restrict__ wb,
                                                        float* __restrict__ out) {
  __shared__ __bf16 As[2][BM * LDA];  // 2 x 20 KB
  __shared__ __bf16 Bs[2][BN * LDB];  // 2 x 10 KB

  const int tid  = threadIdx.x;
  const int lane = tid & 31;
  const int wv   = tid >> 5;   // wave 0..7
  const int wm   = wv & 3;     // wave M index (0..3) -> 64 rows each
  const int wn   = wv >> 2;    // wave N index (0..1) -> 64 cols each
  const int hi   = lane >> 4;  // lane half (0/1)
  const int ln   = lane & 15;

  const int tn0 = blockIdx.x * BN;
  const int tm0 = blockIdx.y * BM;

  v8f acc[4][4] = {};

  // Prologue: stage K-slice 0 into buffer 0.
  stage_tiles(xb, wb, As[0], Bs[0], tm0, tn0, 0, tid);
#if USE_ASYNC
  __builtin_amdgcn_s_wait_asynccnt(0);
#endif
  __syncthreads();

  int buf = 0;
  for (int kt = 0; kt < D_IN; kt += KB) {
    // Stage next K-slice into the other buffer while computing this one.
    if (kt + KB < D_IN)
      stage_tiles(xb, wb, As[buf ^ 1], Bs[buf ^ 1], tm0, tn0, kt + KB, tid);

    // --- fragment loads per CDNA5 16-bit WMMA VGPR layouts ---
    v16bf afrag[4], bfrag[4];
#pragma unroll
    for (int i = 0; i < 4; ++i) {
      // A 16x32: lane = row (M = ln); elems 0-7 -> K = hi*8.., 8-15 -> K = hi*8+16..
      const __bf16* ap = &As[buf][(wm * 64 + i * 16 + ln) * LDA + hi * 8];
      v8bf lo = *(const v8bf*)ap;
      v8bf up = *(const v8bf*)(ap + 16);
      afrag[i] = __builtin_shufflevector(lo, up, 0, 1, 2, 3, 4, 5, 6, 7,
                                         8, 9, 10, 11, 12, 13, 14, 15);
    }
#pragma unroll
    for (int j = 0; j < 4; ++j) {
      // B 32x16: lane = col (N = ln); elems 0-7 -> K = hi*8.., 8-15 -> K = hi*8+16..
      const __bf16* bp = &Bs[buf][(wn * 64 + j * 16 + ln) * LDB + hi * 8];
      v8bf lo = *(const v8bf*)bp;
      v8bf up = *(const v8bf*)(bp + 16);
      bfrag[j] = __builtin_shufflevector(lo, up, 0, 1, 2, 3, 4, 5, 6, 7,
                                         8, 9, 10, 11, 12, 13, 14, 15);
    }

#pragma unroll
    for (int i = 0; i < 4; ++i)
#pragma unroll
      for (int j = 0; j < 4; ++j)
        acc[i][j] = __builtin_amdgcn_wmma_f32_16x16x32_bf16(
            false, afrag[i], false, bfrag[j], (short)0, acc[i][j], false, false);

#if USE_ASYNC
    __builtin_amdgcn_s_wait_asynccnt(0);
#endif
    __syncthreads();
    buf ^= 1;
  }

  // --- epilogue: C/D layout — VGPR r holds row (i*16 + 8*hi + r), col = ln ---
#pragma unroll
  for (int i = 0; i < 4; ++i) {
#pragma unroll
    for (int j = 0; j < 4; ++j) {
      const int row0 = tm0 + wm * 64 + i * 16 + hi * 8;
      const int col  = tn0 + wn * 64 + j * 16 + ln;
      float* op = out + (size_t)row0 * D_OUT + col;
#pragma unroll
      for (int r = 0; r < 8; ++r) op[(size_t)r * D_OUT] = acc[i][j][r];
    }
  }
}

extern "C" void kernel_launch(void* const* d_in, const int* in_sizes, int n_in,
                              void* d_out, int out_size, void* d_ws, size_t ws_size,
                              hipStream_t stream) {
  (void)in_sizes; (void)n_in; (void)out_size; (void)ws_size;
  const float* x  = (const float*)d_in[0];
  const float* wp = (const float*)d_in[1];
  const float* wn = (const float*)d_in[2];
  float* out = (float*)d_out;

  // Workspace layout (requires ~100.7 MB, fits entirely in the 192 MB L2):
  //   [0, 64 MiB)   : xb = bf16(x)           (TOKENS*D_IN*2 bytes)
  //   [64, 96 MiB)  : wb = ternary W as bf16 (D_OUT*D_IN*2 bytes)
  __bf16* xb = (__bf16*)d_ws;
  __bf16* wb = (__bf16*)((char*)d_ws + (size_t)TOKENS * D_IN * 2);

  cvt_x_kernel<<<(TOKENS * D_IN / 8) / 256, 256, 0, stream>>>(x, xb);
  ternarize_kernel<<<(D_OUT * D_IN / 8) / 256, 256, 0, stream>>>(wp, wn, wb);

  dim3 grid(D_OUT / BN, TOKENS / BM);  // 32 x 32 blocks
  tern_gemm_kernel<<<grid, 256, 0, stream>>>(xb, wb, out);
}
